// MHGCL_32246614458470
// MI455X (gfx1250) — matrix operand order repaired and checked by reading, not compiled, not run on hardware
//
#include <hip/hip_runtime.h>
#include <hip/hip_bf16.h>
#include <stdint.h>

typedef __attribute__((ext_vector_type(16))) __bf16 v16bf;
typedef __attribute__((ext_vector_type(8)))  float  v8f;
typedef unsigned short u16;

#define NEWS_N 40000
#define AUG_ELEMS (40000 * 128)

// ---------- helpers ----------
__device__ __forceinline__ u16 f2bf(float f) {
  unsigned u = __float_as_uint(f);
  unsigned r = u + 0x7FFFu + ((u >> 16) & 1u);   // round-to-nearest-even
  return (u16)(r >> 16);
}
__device__ __forceinline__ unsigned enc_f(float f) {   // order-preserving float->uint
  unsigned u = __float_as_uint(f);
  return (u & 0x80000000u) ? ~u : (u | 0x80000000u);
}
__device__ __forceinline__ float dec_f(unsigned u) {
  unsigned v = (u & 0x80000000u) ? (u & 0x7FFFFFFFu) : ~u;
  return __uint_as_float(v);
}

// ---------- conversion kernels ----------
// f32 [N,K] -> bf16 [N,Kp] zero-padded
__global__ void cvt_pad(const float* __restrict__ in, u16* __restrict__ out,
                        int N, int K, int Kp) {
  long t = (long)blockIdx.x * blockDim.x + threadIdx.x;
  if (t >= (long)N * Kp) return;
  int k = (int)(t % Kp);
  long r = t / Kp;
  out[t] = (k < K) ? f2bf(in[r * K + k]) : (u16)0;
}

// f32 [M,128] -> bf16 [M,ldo] starting at column col0
__global__ void cvt_cols(const float* __restrict__ in, u16* __restrict__ out,
                         int M, int ldo, int col0) {
  long t = (long)blockIdx.x * blockDim.x + threadIdx.x;
  if (t >= (long)M * 128) return;
  int c = (int)(t & 127);
  long r = t >> 7;
  out[r * ldo + col0 + c] = f2bf(in[t]);
}

// weight f32 [K,128] -> transposed bf16 [128,Kp] zero-padded
__global__ void wt_cvt(const float* __restrict__ w, u16* __restrict__ out, int K, int Kp) {
  int t = blockIdx.x * blockDim.x + threadIdx.x;
  if (t >= 128 * Kp) return;
  int n = t / Kp, k = t % Kp;
  out[t] = (k < K) ? f2bf(w[(long)k * 128 + n]) : (u16)0;
}

// ---------- WMMA GEMM: C[M,128] = A[M,Kp]bf16 * Bt[128,Kp]^T + bias ----------
__global__ __launch_bounds__(256) void gemm_bf16_n128(
    const u16* __restrict__ A, int M, int Kp,
    const u16* __restrict__ Bt, const float* __restrict__ bias,
    float* __restrict__ Cout, int ldc) {
  const int wave = threadIdx.x >> 5;       // 8 waves -> 8 n-tiles (N=128)
  const int lane = threadIdx.x & 31;
  const int half = lane >> 4;              // lane group 0: lanes 0-15, 1: lanes 16-31
  const int lr   = lane & 15;
  const int m0 = blockIdx.x * 16;
  const int n0 = wave * 16;

  int arow = m0 + lr; if (arow >= M) arow = M - 1;   // clamp; store is guarded
  const u16* Ap = A + (long)arow * Kp;
  const u16* Bp = Bt + (long)(n0 + lr) * Kp + half * 16;

  union Frag { uint4 u[2]; v16bf v; };
  v8f acc = {};
  for (int k = 0; k < Kp; k += 32) {
    Frag a, b;
    // A 16x32 bf16, contiguous k order used here (B matches): lanes 0-15 hold K k..k+15; lanes 16-31 K k+16..k+31
    a.u[0] = *(const uint4*)(Ap + k + half * 16);
    a.u[1] = *(const uint4*)(Ap + k + half * 16 + 8);
    // B 32x16, same contiguous k order as A: lanes 0-15 N=lr K k..k+15; lanes 16-31 K k+16..k+31
    b.u[0] = *(const uint4*)(Bp + k);
    b.u[1] = *(const uint4*)(Bp + k + 8);
    acc = __builtin_amdgcn_wmma_f32_16x16x32_bf16(false, a.v, false, b.v,
                                                  (short)0, acc, false, false);
  }
  const int n = n0 + lr;
  const float bv = bias ? bias[n] : 0.0f;
#pragma unroll
  for (int r = 0; r < 8; ++r) {
    int m = m0 + r + half * 8;             // C/D layout: VGPR r -> M = r (+8 for high lanes)
    if (m < M) Cout[(long)m * ldc + n] = acc[r] + bv;
  }
}

// ---------- GATv2 edge kernels (dst = news), one wave32 per edge ----------
__global__ void init_seg(unsigned* __restrict__ m_enc, float* __restrict__ denom, int n) {
  int t = blockIdx.x * blockDim.x + threadIdx.x;
  if (t < n) { m_enc[t] = 0u; denom[t] = 0.0f; }
}

__global__ void init_aug(float* __restrict__ aug, int n,
                         const float* __restrict__ b0, const float* __restrict__ b1,
                         const float* __restrict__ b2, const float* __restrict__ b3) {
  int t = blockIdx.x * blockDim.x + threadIdx.x;
  if (t >= n) return;
  int c = t & 127;
  float v = b0[c] + b1[c] + b2[c];
  if (b3) v += b3[c];
  aug[t] = v;
}

__global__ void edge_score(const int* __restrict__ ei, int E,
                           const float* __restrict__ xl, const float* __restrict__ xr,
                           const float* __restrict__ att,
                           float* __restrict__ e_arr, unsigned* __restrict__ m_enc) {
  long gid = (long)blockIdx.x * blockDim.x + threadIdx.x;
  int e = (int)(gid >> 5);
  int lane = threadIdx.x & 31;
  if (e >= E) return;
  int s = ei[e], d = ei[E + e];
  float4 xlv = ((const float4*)(xl + (long)s * 128))[lane];
  float4 xrv = ((const float4*)(xr + (long)d * 128))[lane];
  float4 av  = ((const float4*)att)[lane];
  float z, t = 0.0f;
  z = xlv.x + xrv.x; t += (z > 0.0f ? z : 0.2f * z) * av.x;
  z = xlv.y + xrv.y; t += (z > 0.0f ? z : 0.2f * z) * av.y;
  z = xlv.z + xrv.z; t += (z > 0.0f ? z : 0.2f * z) * av.z;
  z = xlv.w + xrv.w; t += (z > 0.0f ? z : 0.2f * z) * av.w;
#pragma unroll
  for (int off = 16; off > 0; off >>= 1) t += __shfl_xor(t, off, 32);
  if (lane == 0) { e_arr[e] = t; atomicMax(m_enc + d, enc_f(t)); }
}

__global__ void edge_alpha(const int* __restrict__ ei, int E,
                           const float* __restrict__ e_arr, const unsigned* __restrict__ m_enc,
                           float* __restrict__ a_arr, float* __restrict__ denom) {
  int e = blockIdx.x * blockDim.x + threadIdx.x;
  if (e >= E) return;
  int d = ei[E + e];
  unsigned mu = m_enc[d];
  float m = (mu == 0u) ? 0.0f : dec_f(mu);
  float a = __expf(e_arr[e] - m);
  a_arr[e] = a;
  atomicAdd(denom + d, a);
}

__global__ void edge_scatter(const int* __restrict__ ei, int E,
                             const float* __restrict__ xl, const float* __restrict__ a_arr,
                             const float* __restrict__ denom, float* __restrict__ out) {
  long gid = (long)blockIdx.x * blockDim.x + threadIdx.x;
  int e = (int)(gid >> 5);
  int lane = threadIdx.x & 31;
  if (e >= E) return;
  int s = ei[e], d = ei[E + e];
  float alpha = a_arr[e] / (denom[d] + 1e-16f);
  float4 xlv = ((const float4*)(xl + (long)s * 128))[lane];
  float* o = out + (long)d * 128 + lane * 4;
  atomicAdd(o + 0, xlv.x * alpha);
  atomicAdd(o + 1, xlv.y * alpha);
  atomicAdd(o + 2, xlv.z * alpha);
  atomicAdd(o + 3, xlv.w * alpha);
}

__global__ void relu_k(float* __restrict__ x, int n) {
  int t = blockIdx.x * blockDim.x + threadIdx.x;
  if (t < n) x[t] = fmaxf(x[t], 0.0f);
}
__global__ void elu_k(float* __restrict__ x, int n) {
  int t = blockIdx.x * blockDim.x + threadIdx.x;
  if (t < n) { float z = x[t]; x[t] = z > 0.0f ? z : (__expf(z) - 1.0f); }
}

// ---------- host orchestration ----------
extern "C" void kernel_launch(void* const* d_in, const int* in_sizes, int n_in,
                              void* d_out, int out_size, void* d_ws, size_t ws_size,
                              hipStream_t stream) {
  (void)n_in; (void)out_size; (void)ws_size;
  char* ws = (char*)d_ws;
  size_t off = 0;
  auto alloc = [&](size_t bytes) { size_t r = off; off += (bytes + 255) & ~(size_t)255; return r; };

  // workspace layout
  u16* news_bf  = (u16*)(ws + alloc((size_t)40000 * 768 * 2));
  u16* ent_bf   = (u16*)(ws + alloc((size_t)80000 * 320 * 2));
  u16* topic_bf = (u16*)(ws + alloc((size_t)1000  * 320 * 2));
  u16* kg_bf    = (u16*)(ws + alloc((size_t)80000 * 320 * 2));
  u16* kg1_bf   = (u16*)(ws + alloc((size_t)80000 * 320 * 2));
  u16* wtA      = (u16*)(ws + alloc((size_t)128 * 768 * 2));
  u16* wtB      = (u16*)(ws + alloc((size_t)128 * 768 * 2));
  float* xl     = (float*)(ws + alloc((size_t)80000 * 128 * 4));
  float* xr     = (float*)(ws + alloc((size_t)40000 * 128 * 4));
  float* e_arr  = (float*)(ws + alloc((size_t)400000 * 4));
  float* a_arr  = (float*)(ws + alloc((size_t)400000 * 4));
  unsigned* m_enc = (unsigned*)(ws + alloc((size_t)40000 * 4));
  float* denom  = (float*)(ws + alloc((size_t)40000 * 4));
  float* aug[3];
  for (int g = 0; g < 3; ++g) aug[g] = (float*)(ws + alloc((size_t)AUG_ELEMS * 4));
  float* hbuf   = (float*)(ws + alloc((size_t)AUG_ELEMS * 4));
  u16* abf      = (u16*)(ws + alloc((size_t)40000 * 384 * 2));

  auto LB = [](long n, int b) { return (unsigned)((n + b - 1) / b); };
  const float* P = nullptr; (void)P;
  auto F  = [&](int i) { return (const float*)d_in[i]; };
  auto I32= [&](int i) { return (const int*)d_in[i]; };

  // stage bf16 node features (Kp multiple of 32)
  cvt_pad<<<LB((long)40000 * 768, 256), 256, 0, stream>>>(F(0), news_bf, 40000, 768, 768);
  cvt_pad<<<LB((long)80000 * 320, 256), 256, 0, stream>>>(F(1), ent_bf,  80000, 300, 320);
  cvt_pad<<<LB((long)1000  * 320, 256), 256, 0, stream>>>(F(2), topic_bf, 1000, 300, 320);
  cvt_pad<<<LB((long)80000 * 320, 256), 256, 0, stream>>>(F(3), kg_bf,   80000, 300, 320);
  cvt_pad<<<LB((long)80000 * 320, 256), 256, 0, stream>>>(F(4), kg1_bf,  80000, 300, 320);

  // group accumulators: init with summed conv biases (params: wl,bl,wr,br,att,bias)
  init_aug<<<LB(AUG_ELEMS, 256), 256, 0, stream>>>(aug[0], AUG_ELEMS, F(23+5),  F(35+5),  F(47+5),  nullptr);
  init_aug<<<LB(AUG_ELEMS, 256), 256, 0, stream>>>(aug[1], AUG_ELEMS, F(71+5),  F(83+5),  F(95+5),  nullptr);
  init_aug<<<LB(AUG_ELEMS, 256), 256, 0, stream>>>(aug[2], AUG_ELEMS, F(119+5), F(131+5), F(143+5), F(155+5));

  // 10 GATv2 instances with dst == 'news'
  struct Inst { const u16* src_bf; int Nsrc; int edge; int p; int group; };
  Inst insts[10] = {
    { ent_bf,   80000,  7,  23, 0 }, { topic_bf, 1000,  9,  35, 0 }, { kg_bf,  80000, 11,  47, 0 },
    { ent_bf,   80000,  7,  71, 1 }, { topic_bf, 1000,  9,  83, 1 }, { kg1_bf, 80000, 13,  95, 1 },
    { ent_bf,   80000,  7, 119, 2 }, { topic_bf, 1000,  9, 131, 2 }, { kg_bf,  80000, 11, 143, 2 },
    { kg1_bf,   80000, 13, 155, 2 },
  };

  for (int i = 0; i < 10; ++i) {
    const Inst& t = insts[i];
    const int E = in_sizes[t.edge] / 2;
    const int* ei = I32(t.edge);
    // wl:[300,128] -> wtA[128,320] ; wr:[768,128] -> wtB[128,768]
    wt_cvt<<<LB(128 * 320, 256), 256, 0, stream>>>(F(t.p + 0), wtA, 300, 320);
    wt_cvt<<<LB(128 * 768, 256), 256, 0, stream>>>(F(t.p + 2), wtB, 768, 768);
    // xl = x_src @ wl + bl ; xr = x_news @ wr + br
    gemm_bf16_n128<<<(t.Nsrc + 15) / 16, 256, 0, stream>>>(t.src_bf, t.Nsrc, 320, wtA, F(t.p + 1), xl, 128);
    gemm_bf16_n128<<<(NEWS_N + 15) / 16, 256, 0, stream>>>(news_bf, NEWS_N, 768, wtB, F(t.p + 3), xr, 128);
    // segment softmax + weighted scatter
    init_seg<<<LB(NEWS_N, 256), 256, 0, stream>>>(m_enc, denom, NEWS_N);
    edge_score<<<LB((long)E * 32, 256), 256, 0, stream>>>(ei, E, xl, xr, F(t.p + 4), e_arr, m_enc);
    edge_alpha<<<LB(E, 256), 256, 0, stream>>>(ei, E, e_arr, m_enc, a_arr, denom);
    edge_scatter<<<LB((long)E * 32, 256), 256, 0, stream>>>(ei, E, xl, a_arr, denom, aug[t.group]);
  }
  for (int g = 0; g < 3; ++g)
    relu_k<<<LB(AUG_ELEMS, 256), 256, 0, stream>>>(aug[g], AUG_ELEMS);

  // proj: elu(aug @ fc1 + b1) @ fc2 + b2  -> d_out sections 0..2
  float* outf = (float*)d_out;
  wt_cvt<<<LB(128 * 128, 256), 256, 0, stream>>>(F(179), wtA, 128, 128);  // fc1_w
  wt_cvt<<<LB(128 * 128, 256), 256, 0, stream>>>(F(180), wtB, 128, 128);  // fc2_w
  for (int g = 0; g < 3; ++g) {
    cvt_cols<<<LB(AUG_ELEMS, 256), 256, 0, stream>>>(aug[g], abf, 40000, 128, 0);
    gemm_bf16_n128<<<(NEWS_N + 15) / 16, 256, 0, stream>>>(abf, NEWS_N, 128, wtA, F(182), hbuf, 128);
    elu_k<<<LB(AUG_ELEMS, 256), 256, 0, stream>>>(hbuf, AUG_ELEMS);
    cvt_cols<<<LB(AUG_ELEMS, 256), 256, 0, stream>>>(hbuf, abf, 40000, 128, 0);
    gemm_bf16_n128<<<(NEWS_N + 15) / 16, 256, 0, stream>>>(abf, NEWS_N, 128, wtB, F(183),
                                                           outf + (size_t)g * AUG_ELEMS, 128);
  }
  // news = concat(a1,a2,a3) @ lin2_w + lin2_b -> d_out section 3
  for (int g = 0; g < 3; ++g)
    cvt_cols<<<LB(AUG_ELEMS, 256), 256, 0, stream>>>(outf + (size_t)g * AUG_ELEMS, abf, 40000, 384, g * 128);
  wt_cvt<<<LB(128 * 384, 256), 256, 0, stream>>>(F(181), wtA, 384, 384);  // lin2_w [384,128]
  gemm_bf16_n128<<<(NEWS_N + 15) / 16, 256, 0, stream>>>(abf, NEWS_N, 384, wtA, F(184),
                                                         outf + (size_t)3 * AUG_ELEMS, 128);
}
